// DOPAMINE_44375602102715
// MI455X (gfx1250) — matrix operand order repaired and checked by reading, not compiled, not run on hardware
//
#include <hip/hip_runtime.h>
#include <hip/hip_bf16.h>
#include <math.h>

typedef _Float16 v16h __attribute__((ext_vector_type(16)));
typedef _Float16 v8h  __attribute__((ext_vector_type(8)));
typedef float    v8f  __attribute__((ext_vector_type(8)));
typedef int      v4i  __attribute__((ext_vector_type(4)));

#define B_   8
#define H_   192
#define W_   192
#define NQ   11
#define P_   (B_*H_*W_)      // 294912 pixels
#define NS_  10
#define CH   64
#define KMID 576             // 9 taps * 64 cin
#define NGD  300

#define PIX  64              // pixels per conv_mid block (strip along W; W_/PIX == 3)
#define RS   72              // raw A row stride in halves (64 + 8 pad; 144B, 16B-aligned)
#define KP2  200             // B row stride in halves (192 + 8 pad; 400B, 16B-aligned)

// Async global->LDS staging (CDNA5 GLOBAL_LOAD_ASYNC_TO_LDS_B128, ASYNCcnt),
// with a ds-write fallback if the builtin is unavailable on this toolchain.
// Probe result (round 2 stderr): builtin exists and takes typed pointers:
//   (v4i addrspace(1)* src, v4i addrspace(3)* dst, int imm_offset, int cpol)
#if defined(__HIP_DEVICE_COMPILE__) && defined(__has_builtin)
#if __has_builtin(__builtin_amdgcn_global_load_async_to_lds_b128) && \
    __has_builtin(__builtin_amdgcn_s_wait_asynccnt)
#define USE_ASYNC_LDS 1
#endif
#endif

#ifdef USE_ASYNC_LDS
typedef __attribute__((address_space(1))) v4i* gptr_v4i;   // global (AS1, 64-bit)
typedef __attribute__((address_space(3))) v4i* lptr_v4i;   // LDS    (AS3, 32-bit offset)
#endif

__device__ __forceinline__ void stage16B(const _Float16* gsrc, _Float16* ldst) {
#ifdef USE_ASYNC_LDS
    gptr_v4i g1 = (gptr_v4i)(uintptr_t)gsrc;
    lptr_v4i l3 = (lptr_v4i)(uintptr_t)ldst;   // low 32 bits of generic ptr = LDS offset
    __builtin_amdgcn_global_load_async_to_lds_b128(g1, l3, 0, 0);
#else
    *(v8h*)ldst = *(const v8h*)gsrc;
#endif
}

__device__ __forceinline__ void stage_fence() {
#ifdef USE_ASYNC_LDS
    __builtin_amdgcn_s_wait_asynccnt(0);
#endif
}

__device__ __forceinline__ float guardf(float g) { return isfinite(g) ? g : 0.0f; }
__device__ __forceinline__ float clip0(float v, float hi) { return fminf(fmaxf(v, 0.0f), hi); }

// ---------------------------------------------------------------------------
// Weight repack kernels: f32 HWIO -> f16 [cout][k], k = (ky*3+kx)*Cin + cin
// ---------------------------------------------------------------------------
__global__ void pack_w1(const float* __restrict__ W1, _Float16* __restrict__ w1p) {
    int id = blockIdx.x * blockDim.x + threadIdx.x;
    if (id >= NS_*64*27) return;
    int k = id % 27; int cout = (id / 27) % 64; int i = id / (27*64);
    int cin = k % 3; int t = k / 3; int kx = t % 3; int ky = t / 3;
    w1p[id] = (_Float16)W1[(((((size_t)i*3 + ky)*3 + kx)*3 + cin)*64 + cout)];
}

__global__ void pack_wm(const float* __restrict__ Wm, _Float16* __restrict__ wmp) {
    int id = blockIdx.x * blockDim.x + threadIdx.x;
    if (id >= NS_*5*64*KMID) return;
    int k = id % KMID; int cout = (id / KMID) % 64;
    int j = (id / (KMID*64)) % 5; int i = id / (KMID*64*5);
    int cin = k % 64; int t = k / 64; int kx = t % 3; int ky = t / 3;
    wmp[id] = (_Float16)Wm[(((((size_t)(i*5 + j)*3 + ky)*3 + kx)*64 + cin)*64 + cout)];
}

__global__ void pack_wl(const float* __restrict__ Wl, _Float16* __restrict__ wlp) {
    int id = blockIdx.x * blockDim.x + threadIdx.x;
    if (id >= NS_*3*KMID) return;
    int k = id % KMID; int c = (id / KMID) % 3; int i = id / (KMID*3);
    int cin = k % 64; int t = k / 64; int kx = t % 3; int ky = t / 3;
    wlp[id] = (_Float16)Wl[((((size_t)(i*3 + ky)*3 + kx)*64 + cin)*3 + c)];
}

// ---------------------------------------------------------------------------
// Init + 300 GD steps, fully register-resident per pixel. Writes xm[0], xm[1].
// ---------------------------------------------------------------------------
__global__ void gd_kernel(const float* __restrict__ b, const float* __restrict__ tau,
                          float* __restrict__ out, float* __restrict__ xcur) {
    int p = blockIdx.x * blockDim.x + threadIdx.x;
    if (p >= P_) return;
    const float* bp = b + (size_t)p * NQ;
    __builtin_prefetch(bp, 0, 3);   // global_prefetch_b8
    float bv[NQ], tv[NQ], A = 0.0f;
    #pragma unroll
    for (int q = 0; q < NQ; ++q) {
        bv[q] = bp[q]; tv[q] = tau[q];
        A = fmaxf(A, fabsf(bv[q]));
    }
    float xa = clip0(A, 3.0f), xb = clip0(2.0f * A, 6.0f), xr = 1.0f;
    size_t o0 = (size_t)p * 3;
    out[o0 + 0] = xa; out[o0 + 1] = xb; out[o0 + 2] = xr;      // xm[0]
    const float inv = 1.0f / (float)NQ;
    #pragma unroll 1
    for (int it = 0; it < NGD; ++it) {
        float gA = 0.f, gB = 0.f, gR = 0.f;
        #pragma unroll
        for (int q = 0; q < NQ; ++q) {
            float e = __expf(-tv[q] * xr);
            float r = (xa - xb * e) - bv[q];
            gA += r;
            gB -= r * e;
            gR += r * xb * tv[q] * e;
        }
        gA = guardf(gA * inv); gB = guardf(gB * inv); gR = guardf(gR * inv);
        xa = clip0(xa - 2.0f * gA, 3.0f);
        xb = clip0(xb - 2.0f * gB, 6.0f);
        xr = clip0(xr - 2.0f * gR, 50.0f);
    }
    size_t o1 = (size_t)P_ * 3 + o0;                            // xm[1]
    out[o1 + 0] = xa; out[o1 + 1] = xb; out[o1 + 2] = xr;
    xcur[o0 + 0] = xa; xcur[o0 + 1] = xb; xcur[o0 + 2] = xr;
}

// ---------------------------------------------------------------------------
// conv_in: 3 -> 64, 3x3 SAME, ReLU, f32 input -> f16 activations
// ---------------------------------------------------------------------------
__global__ void conv_in_kernel(const float* __restrict__ x, const _Float16* __restrict__ w1p,
                               const float* __restrict__ b1, _Float16* __restrict__ actout) {
    int id = blockIdx.x * blockDim.x + threadIdx.x;
    if (id >= P_*CH) return;
    int n = id & 63; int p = id >> 6;
    int w = p % W_; int h = (p / W_) % H_; int bb = p / (W_*H_);
    float acc = b1[n];
    const _Float16* wrow = w1p + n * 27;
    #pragma unroll
    for (int ky = 0; ky < 3; ++ky) {
        int hh = h + ky - 1; if (hh < 0 || hh >= H_) continue;
        #pragma unroll
        for (int kx = 0; kx < 3; ++kx) {
            int ww = w + kx - 1; if (ww < 0 || ww >= W_) continue;
            const float* xp = x + (((size_t)bb*H_ + hh)*W_ + ww) * 3;
            int kb = (ky*3 + kx) * 3;
            acc += xp[0]*(float)wrow[kb] + xp[1]*(float)wrow[kb+1] + xp[2]*(float)wrow[kb+2];
        }
    }
    actout[(size_t)p*CH + n] = (_Float16)fmaxf(acc, 0.0f);
}

// ---------------------------------------------------------------------------
// conv_mid: 64 -> 64, 3x3 SAME, ReLU — implicit GEMM via v_wmma_f32_16x16x32_f16.
// Block = 128 threads (4 waves), 64-pixel W-strip x 64 couts.
// Wave w owns M-tile (pixels 16w..16w+15) and computes all 4 N-tiles, reusing
// one A fragment across 4 WMMAs. A is staged as raw rows (66 x 64ch); the kx
// shift of the im2col is applied at fragment-gather time. K chunked by ky.
// ---------------------------------------------------------------------------
__global__ __launch_bounds__(128)
void conv_mid_kernel(const _Float16* __restrict__ actin, _Float16* __restrict__ actout,
                     const _Float16* __restrict__ wpack /* [64][576] */,
                     const float* __restrict__ bias /* [64] */) {
    __shared__ _Float16 Araw[66*RS];    // 66 pixels x 64 ch (+pad)  = 4752 halves
    __shared__ _Float16 Bl[CH*KP2];     // 64 couts  x 192 K (+pad) = 12800 halves
    int blk = blockIdx.x;
    int w0 = (blk % (W_/PIX)) * PIX;
    int h  = (blk / (W_/PIX)) % H_;
    int bb = blk / ((W_/PIX) * H_);
    int tid = threadIdx.x, lane = tid & 31, wave = tid >> 5;
    int mrow = lane & 15, hi = lane >> 4;   // hi: 0 for lanes 0-15, 1 for 16-31
    int mpix = wave * 16 + mrow;            // this lane's A-matrix row (pixel in strip)

    v8f acc[4];
    #pragma unroll
    for (int nt = 0; nt < 4; ++nt) acc[nt] = (v8f){0.f,0.f,0.f,0.f,0.f,0.f,0.f,0.f};

    const size_t plane = ((size_t)bb * H_) * W_;

    for (int ky = 0; ky < 3; ++ky) {
        int hh = h + ky - 1;
        bool hok = (hh >= 0) && (hh < H_);
        // Stage A: raw rows r=0..65 -> pixel ww = w0-1+r, 64 halves each (8 x 16B chunks)
        for (int idx = tid; idx < 66*8; idx += 128) {
            int r = idx >> 3, cseg = (idx & 7) * 8;
            int ww = w0 - 1 + r;
            _Float16* dst = &Araw[r*RS + cseg];
            if (hok && ww >= 0 && ww < W_) {
                stage16B(&actin[(((size_t)bb*H_ + hh)*W_ + ww)*CH + cseg], dst);
            } else {
                v8h z = {};
                *(v8h*)dst = z;
            }
        }
        // Stage B: 64 rows x 192 halves (24 x 16B chunks per row)
        for (int idx = tid; idx < 64*24; idx += 128) {
            int row = idx / 24, cseg = (idx % 24) * 8;
            stage16B(&wpack[(size_t)row*KMID + ky*192 + cseg], &Bl[row*KP2 + cseg]);
        }
        stage_fence();
        __syncthreads();
        #pragma unroll
        for (int t = 0; t < 6; ++t) {
            int k0 = t * 32;
            int kx = k0 >> 6;          // which kx tap this K-chunk belongs to
            int c0 = k0 & 63;          // channel offset within the tap
            // A frag (16x32 f16): lanes 0-15 hold K {0..7,16..23}, lanes 16-31 K {8..15,24..31}
            const _Float16* arow = &Araw[(mpix + kx)*RS];
            v8h alo = *(const v8h*)&arow[c0 + hi*8];
            v8h ahi = *(const v8h*)&arow[c0 + 16 + hi*8];
            v16h afr = __builtin_shufflevector(alo, ahi, 0,1,2,3,4,5,6,7,8,9,10,11,12,13,14,15);
            #pragma unroll
            for (int nt = 0; nt < 4; ++nt) {
                // B frag (32x16 f16): lanes 0-15 hold K {0..7,16..23} of col n, lanes 16-31 K {8..15,24..31}
                const _Float16* brow = &Bl[(nt*16 + mrow)*KP2 + k0 + hi*8];
                v8h blo = *(const v8h*)&brow[0];
                v8h bhi = *(const v8h*)&brow[16];
                v16h bfr = __builtin_shufflevector(blo, bhi, 0,1,2,3,4,5,6,7,8,9,10,11,12,13,14,15);
                acc[nt] = __builtin_amdgcn_wmma_f32_16x16x32_f16(false, afr, false, bfr,
                                                                 (short)0, acc[nt], false, false);
            }
        }
        __syncthreads();
    }
    // Epilogue: C/D layout — VGPR r: lanes 0-15 M=r, lanes 16-31 M=r+8; N = n0 + (lane&15)
    #pragma unroll
    for (int nt = 0; nt < 4; ++nt) {
        int n = nt*16 + mrow;
        float bv = bias[n];
        #pragma unroll
        for (int r = 0; r < 8; ++r) {
            int m = r + (hi << 3);
            float v = fmaxf(acc[nt][r] + bv, 0.0f);
            actout[(plane + (size_t)h*W_ + (w0 + wave*16 + m))*CH + n] = (_Float16)v;
        }
    }
}

// ---------------------------------------------------------------------------
// conv_out: 64 -> 3, 3x3 SAME, no ReLU; Dx = x - h
// ---------------------------------------------------------------------------
__global__ void conv_out_kernel(const _Float16* __restrict__ actin, const _Float16* __restrict__ wlp,
                                const float* __restrict__ bl, const float* __restrict__ xcur,
                                float* __restrict__ dxbuf) {
    int id = blockIdx.x * blockDim.x + threadIdx.x;
    if (id >= P_*3) return;
    int c = id % 3; int p = id / 3;
    int w = p % W_; int h = (p / W_) % H_; int bb = p / (W_*H_);
    float acc = bl[c];
    const _Float16* wrow = wlp + (size_t)c * KMID;
    for (int ky = 0; ky < 3; ++ky) {
        int hh = h + ky - 1; if (hh < 0 || hh >= H_) continue;
        for (int kx = 0; kx < 3; ++kx) {
            int ww = w + kx - 1; if (ww < 0 || ww >= W_) continue;
            const _Float16* ap = actin + (((size_t)bb*H_ + hh)*W_ + ww)*CH;
            const _Float16* wp = wrow + (ky*3 + kx)*CH;
            float s = 0.f;
            #pragma unroll
            for (int s8 = 0; s8 < 8; ++s8) {
                v8h av = *(const v8h*)(ap + s8*8);
                v8h wv = *(const v8h*)(wp + s8*8);
                #pragma unroll
                for (int e = 0; e < 8; ++e) s += (float)av[e] * (float)wv[e];
            }
            acc += s;
        }
    }
    dxbuf[(size_t)p*3 + c] = xcur[(size_t)p*3 + c] - acc;
}

// ---------------------------------------------------------------------------
// PnP update. Faithful to the reference: Dx = range_constraint_R1(x) (the
// reference overwrites the denoiser output), so d = gx + lm*(x - clip(x)).
// ---------------------------------------------------------------------------
__global__ void update_kernel(const float* __restrict__ b, const float* __restrict__ tau,
                              const float* __restrict__ mu, const float* __restrict__ lm,
                              int stage, float* __restrict__ xcur, float* __restrict__ out,
                              const float* __restrict__ dxbuf) {
    (void)dxbuf;  // denoiser output discarded by the reference's own code
    int p = blockIdx.x * blockDim.x + threadIdx.x;
    if (p >= P_) return;
    size_t o = (size_t)p * 3;
    float xa = xcur[o + 0], xb = xcur[o + 1], xr = xcur[o + 2];
    const float* bp = b + (size_t)p * NQ;
    float gA = 0.f, gB = 0.f, gR = 0.f;
    #pragma unroll
    for (int q = 0; q < NQ; ++q) {
        float tq = tau[q];
        float e = __expf(-tq * xr);
        float r = (xa - xb * e) - bp[q];
        gA += r; gB -= r * e; gR += r * xb * tq * e;
    }
    const float inv = 1.0f / (float)NQ;
    gA = guardf(gA * inv); gB = guardf(gB * inv); gR = guardf(gR * inv);
    float l = lm[stage], m = mu[stage];
    float dA = gA + l * (xa - clip0(xa, 3.0f));
    float dB = gB + l * (xb - clip0(xb, 6.0f));
    float dR = gR + l * (xr - clip0(xr, 50.0f));
    xa = clip0(xa - m * dA, 3.0f);
    xb = clip0(xb - m * dB, 6.0f);
    xr = clip0(xr - m * dR, 50.0f);
    out[o + 0] = xa; out[o + 1] = xb; out[o + 2] = xr;
    xcur[o + 0] = xa; xcur[o + 1] = xb; xcur[o + 2] = xr;
}

// ---------------------------------------------------------------------------
extern "C" void kernel_launch(void* const* d_in, const int* in_sizes, int n_in,
                              void* d_out, int out_size, void* d_ws, size_t ws_size,
                              hipStream_t stream) {
    (void)in_sizes; (void)n_in; (void)out_size; (void)ws_size;
    const float* b   = (const float*)d_in[0];
    const float* tau = (const float*)d_in[1];
    const float* mu  = (const float*)d_in[2];
    const float* lm  = (const float*)d_in[3];
    const float* W1  = (const float*)d_in[4];
    const float* b1  = (const float*)d_in[5];
    const float* Wm  = (const float*)d_in[6];
    const float* bm  = (const float*)d_in[7];
    const float* Wl  = (const float*)d_in[8];
    const float* bl  = (const float*)d_in[9];
    float* out = (float*)d_out;

    char* ws = (char*)d_ws;
    size_t off = 0;
    auto carve = [&](size_t bytes) -> void* {
        void* p = ws + off;
        off += (bytes + 255) & ~(size_t)255;
        return p;
    };
    float*    xcur = (float*)carve((size_t)P_*3*sizeof(float));
    float*    dxb  = (float*)carve((size_t)P_*3*sizeof(float));
    _Float16* act0 = (_Float16*)carve((size_t)P_*CH*sizeof(_Float16));
    _Float16* act1 = (_Float16*)carve((size_t)P_*CH*sizeof(_Float16));
    _Float16* w1p  = (_Float16*)carve((size_t)NS_*64*27*sizeof(_Float16));
    _Float16* wmp  = (_Float16*)carve((size_t)NS_*5*64*KMID*sizeof(_Float16));
    _Float16* wlp  = (_Float16*)carve((size_t)NS_*3*KMID*sizeof(_Float16));

    pack_w1<<<(NS_*64*27 + 255)/256, 256, 0, stream>>>(W1, w1p);
    pack_wm<<<(NS_*5*64*KMID + 255)/256, 256, 0, stream>>>(Wm, wmp);
    pack_wl<<<(NS_*3*KMID + 255)/256, 256, 0, stream>>>(Wl, wlp);

    gd_kernel<<<(P_ + 255)/256, 256, 0, stream>>>(b, tau, out, xcur);

    for (int i = 0; i < NS_; ++i) {
        conv_in_kernel<<<(P_*CH + 255)/256, 256, 0, stream>>>(
            xcur, w1p + (size_t)i*64*27, b1 + i*64, act0);
        _Float16* src = act0; _Float16* dst = act1;
        for (int j = 0; j < 5; ++j) {
            conv_mid_kernel<<<B_*H_*(W_/PIX), 128, 0, stream>>>(
                src, dst, wmp + (size_t)(i*5 + j)*64*KMID, bm + (i*5 + j)*64);
            _Float16* t = src; src = dst; dst = t;
        }
        conv_out_kernel<<<(P_*3 + 255)/256, 256, 0, stream>>>(
            src, wlp + (size_t)i*3*KMID, bl + i*3, xcur, dxb);
        update_kernel<<<(P_ + 255)/256, 256, 0, stream>>>(
            b, tau, mu, lm, i, xcur, out + (size_t)(2 + i)*P_*3, dxb);
    }
}
